// StochasticLSTMCell_5145370820729
// MI455X (gfx1250) — matrix operand order repaired and checked by reading, not compiled, not run on hardware
//
#include <hip/hip_runtime.h>

#define T_DIM 512
#define B_DIM 128
#define I_DIM 256
#define H_DIM 256
#define GATES 4
#define ROWS_PER_WAVE 4

typedef __attribute__((ext_vector_type(16))) __bf16 v16bf;
typedef __attribute__((ext_vector_type(8)))  float  v8f;
typedef int vint4 __attribute__((vector_size(4 * sizeof(int))));

#define AS1 __attribute__((address_space(1)))
#define AS3 __attribute__((address_space(3)))

#if defined(__has_builtin)
#if __has_builtin(__builtin_amdgcn_global_load_async_to_lds_b128)
#define USE_ASYNC_LDS 1
#endif
#endif

__device__ __forceinline__ void wait_async0() {
#if __has_builtin(__builtin_amdgcn_s_wait_asynccnt)
    __builtin_amdgcn_s_wait_asynccnt(0);
#else
    asm volatile("s_wait_asynccnt 0x0" ::: "memory");
#endif
}

__device__ __forceinline__ float sigmoidf_(float x) {
    return 1.0f / (1.0f + __expf(-x));
}

// Load 8 f32 activations * 8 f32 masks -> bf16 into fragment slots [OFF, OFF+8)
#define CVT8M(d, OFF, xp, mp) do {                                         \
    float4 _a0 = *(const float4*)(xp);                                     \
    float4 _a1 = *(const float4*)((xp) + 4);                               \
    float4 _b0 = *(const float4*)(mp);                                     \
    float4 _b1 = *(const float4*)((mp) + 4);                               \
    d[OFF+0] = (__bf16)(_a0.x * _b0.x); d[OFF+1] = (__bf16)(_a0.y * _b0.y);\
    d[OFF+2] = (__bf16)(_a0.z * _b0.z); d[OFF+3] = (__bf16)(_a0.w * _b0.w);\
    d[OFF+4] = (__bf16)(_a1.x * _b1.x); d[OFF+5] = (__bf16)(_a1.y * _b1.y);\
    d[OFF+6] = (__bf16)(_a1.z * _b1.z); d[OFF+7] = (__bf16)(_a1.w * _b1.w);\
} while (0)

// Load 16 contiguous f32 weights -> bf16 B fragment (lane = output column)
#define CVT16(d, wp) do {                                                  \
    float4 _c0 = *(const float4*)(wp);       float4 _c1 = *(const float4*)((wp) + 4);  \
    float4 _c2 = *(const float4*)((wp) + 8); float4 _c3 = *(const float4*)((wp) + 12); \
    d[0]  = (__bf16)_c0.x; d[1]  = (__bf16)_c0.y; d[2]  = (__bf16)_c0.z; d[3]  = (__bf16)_c0.w; \
    d[4]  = (__bf16)_c1.x; d[5]  = (__bf16)_c1.y; d[6]  = (__bf16)_c1.z; d[7]  = (__bf16)_c1.w; \
    d[8]  = (__bf16)_c2.x; d[9]  = (__bf16)_c2.y; d[10] = (__bf16)_c2.z; d[11] = (__bf16)_c2.w; \
    d[12] = (__bf16)_c3.x; d[13] = (__bf16)_c3.y; d[14] = (__bf16)_c3.z; d[15] = (__bf16)_c3.w; \
} while (0)

// ---------------------------------------------------------------------------
// Concrete-dropout mask: z = (1 - sigmoid((logit(p) + logit(u)) / temp)) / (1-p)
// ---------------------------------------------------------------------------
__global__ void mask_kernel(const float* __restrict__ u, float* __restrict__ z, int n) {
    int i = blockIdx.x * blockDim.x + threadIdx.x;
    if (i >= n) return;
    const float EPS = 1e-7f, TEMP = 0.1f, P = 0.25f;
    float logit_p = __logf(P + EPS) - __logf(1.0f - P + EPS);
    float uu = u[i];
    float logit_u = __logf(uu + EPS) - __logf(1.0f - uu + EPS);
    float s = sigmoidf_((logit_p + logit_u) * (1.0f / TEMP));
    z[i] = (1.0f - s) * (1.0f / (1.0f - P));
}

__global__ void zero_kernel(float* __restrict__ p, int n) {
    int i = blockIdx.x * blockDim.x + threadIdx.x;
    if (i < n) p[i] = 0.0f;
}

// ---------------------------------------------------------------------------
// x_pre[t,b,g,h] = sum_i (input[t,b,i]*zx[g,b,i]) * W[g,h,i] + Wb[g,h]
// Workgroup owns a (gate, 16-col) strip: the contiguous 16KB W tile is staged
// to LDS once (async global->LDS), each wave converts its 8 B fragments once
// into VGPRs, then sweeps ROWS_PER_WAVE row tiles reusing them.
// grid = (H/16, T*B/16/(8*ROWS_PER_WAVE), GATES), block = 256 (8 waves).
// ---------------------------------------------------------------------------
__global__ void __launch_bounds__(256) xpre_gemm_kernel(
    const float* __restrict__ input, const float* __restrict__ zx,
    const float* __restrict__ W, const float* __restrict__ Wb,
    float* __restrict__ xpre)
{
    __shared__ float sW[16 * I_DIM];            // 16 KB weight tile

    const int lane  = threadIdx.x & 31;
    const int wave  = threadIdx.x >> 5;
    const int g     = blockIdx.z;
    const int col0  = blockIdx.x * 16;
    const int col   = col0 + (lane & 15);
    const int koff  = (lane >> 4) ? 16 : 0;     // A-fragment half-K offset
    const int bkoff = (lane >> 4) ? 16 : 0;     // B-fragment half-K offset
    const int crow  = (lane >> 4) ? 8  : 0;     // C-fragment row offset

    // Stage W tile: rows col0..col0+15 of W[g] are contiguous (16 KB).
    {
        const float* gW = W + ((size_t)g * H_DIM + col0) * I_DIM;
#ifdef USE_ASYNC_LDS
        for (int j = threadIdx.x; j < 16 * I_DIM / 4; j += 256) {
            AS1 vint4* gsrc = (AS1 vint4*)(gW) + j;
            AS3 vint4* ldst = (AS3 vint4*)(sW) + j;
            __builtin_amdgcn_global_load_async_to_lds_b128(gsrc, ldst, 0, 0);
        }
        wait_async0();
#else
        const float4* src = (const float4*)gW;
        float4* dst = (float4*)sW;
        for (int j = threadIdx.x; j < 16 * I_DIM / 4; j += 256)
            dst[j] = src[j];
#endif
    }
    __syncthreads();

    // Build the 8 K-slab B fragments once; hold in VGPRs for all row tiles.
    v16bf bf[8];
#pragma unroll
    for (int kk = 0; kk < 8; ++kk) {
        const float* p = sW + (lane & 15) * I_DIM + kk * 32 + bkoff;
        CVT16(bf[kk], p);
    }

    const float  wb    = Wb[g * H_DIM + col];
    const float* mrow0 = zx + (size_t)g * B_DIM * I_DIM;

    for (int r = 0; r < ROWS_PER_WAVE; ++r) {
        const int rowTile = (blockIdx.y * 8 + wave) * ROWS_PER_WAVE + r;
        const int rowBase = rowTile * 16;
        const int arow    = rowBase + (lane & 15);
        const int b       = arow & (B_DIM - 1);
        const float* xrow = input + (size_t)arow * I_DIM;
        const float* mrow = mrow0 + (size_t)b * I_DIM;

        // Speculative prefetch of the next row tile's activations.
        __builtin_prefetch(xrow + 16 * I_DIM, 0, 1);

        v8f acc = {};
#pragma unroll
        for (int kk = 0; kk < 8; ++kk) {
            const int k0 = kk * 32;
            v16bf a;
            CVT8M(a, 0, xrow + k0 + koff,      mrow + k0 + koff);
            CVT8M(a, 8, xrow + k0 + 8 + koff, mrow + k0 + 8 + koff);
            acc = __builtin_amdgcn_wmma_f32_16x16x32_bf16(
                false, a, false, bf[kk], (short)0, acc, false, false);
        }
#pragma unroll
        for (int rr8 = 0; rr8 < 8; ++rr8) {
            int rr = rowBase + rr8 + crow;
            xpre[(size_t)rr * (GATES * H_DIM) + g * H_DIM + col] = acc[rr8] + wb;
        }
    }
}

// ---------------------------------------------------------------------------
// One LSTM timestep, fully fused. grid = 16 (htiles), block = 128 (4 waves,
// wave == gate). Each wave converts its 8 U fragments once into VGPRs, then
// loops over all 8 batch tiles: seed acc with x_pre+Ub, 8 WMMAs against the
// held U fragments, stage the s-tile in LDS, fused gate/c/h update.
// ---------------------------------------------------------------------------
__global__ void __launch_bounds__(128) lstm_step_kernel(
    const float* __restrict__ xpre_t,   // x_pre + t*B*4*H
    const float* __restrict__ zh,
    const float* __restrict__ U, const float* __restrict__ Ub,
    const float* __restrict__ h_in,
    float* __restrict__ h_out,
    float* __restrict__ c_buf,
    float* __restrict__ hn_t,           // d_out + t*B*H
    float* __restrict__ ht_out,         // final-h dest (last step) or null
    float* __restrict__ ct_out)         // final-c dest (last step) or null
{
    __shared__ float sld[GATES * 16 * 16];

    const int lane  = threadIdx.x & 31;
    const int g     = threadIdx.x >> 5;     // wave index == gate
    const int htile = blockIdx.x;
    const int col   = htile * 16 + (lane & 15);
    const int koff  = (lane >> 4) ? 16 : 0;
    const int bkoff = (lane >> 4) ? 16 : 0;
    const int crow  = (lane >> 4) ? 8  : 0;

    // Recurrent-weight fragments: converted once, reused for all batch tiles.
    const float* urow = U + ((size_t)g * H_DIM + col) * H_DIM;
    v16bf uf[8];
#pragma unroll
    for (int kk = 0; kk < 8; ++kk)
        CVT16(uf[kk], urow + kk * 32 + bkoff);

    const float ub = Ub[g * H_DIM + col];

    for (int bt = 0; bt < B_DIM / 16; ++bt) {
        const int rowBase = bt * 16;
        const int arow    = rowBase + (lane & 15);
        const float* hrow = h_in + (size_t)arow * H_DIM;
        const float* mrow = zh + ((size_t)g * B_DIM + arow) * H_DIM;

        // Seed accumulator with x_pre + Ub in the C/D register layout.
        v8f acc;
#pragma unroll
        for (int r = 0; r < 8; ++r) {
            int rr = rowBase + r + crow;
            acc[r] = xpre_t[(size_t)rr * (GATES * H_DIM) + g * H_DIM + col] + ub;
        }

#pragma unroll
        for (int kk = 0; kk < 8; ++kk) {
            const int k0 = kk * 32;
            v16bf a;
            CVT8M(a, 0, hrow + k0 + koff,      mrow + k0 + koff);
            CVT8M(a, 8, hrow + k0 + 8 + koff, mrow + k0 + 8 + koff);
            acc = __builtin_amdgcn_wmma_f32_16x16x32_bf16(
                false, a, false, uf[kk], (short)0, acc, false, false);
        }

#pragma unroll
        for (int r = 0; r < 8; ++r)
            sld[g * 256 + (r + crow) * 16 + (lane & 15)] = acc[r];

        __syncthreads();

        // 128 threads handle the 256 tile elements (2 each).
#pragma unroll
        for (int e = threadIdx.x; e < 256; e += 128) {
            int lr = e >> 4, lc = e & 15;
            int bidx = rowBase + lr;
            int hidx = htile * 16 + lc;
            float si = sld[0 * 256 + e];
            float sf = sld[1 * 256 + e];
            float so = sld[2 * 256 + e];
            float sg = sld[3 * 256 + e];
            float ig = sigmoidf_(si);
            float fg = sigmoidf_(sf);
            float og = sigmoidf_(so);
            float gg = tanhf(sg);
            size_t idx = (size_t)bidx * H_DIM + hidx;
            float cn = fg * c_buf[idx] + ig * gg;
            float hn = og * tanhf(cn);
            c_buf[idx] = cn;
            h_out[idx] = hn;
            hn_t[idx]  = hn;
            if (ht_out) { ht_out[idx] = hn; ct_out[idx] = cn; }
        }
        __syncthreads();   // protect sld before next batch tile
    }
}

// ---------------------------------------------------------------------------
extern "C" void kernel_launch(void* const* d_in, const int* in_sizes, int n_in,
                              void* d_out, int out_size, void* d_ws, size_t ws_size,
                              hipStream_t stream)
{
    const float* input = (const float*)d_in[0];
    const float* ux    = (const float*)d_in[1];
    const float* uh    = (const float*)d_in[2];
    const float* W     = (const float*)d_in[3];
    const float* Wb    = (const float*)d_in[4];
    const float* U     = (const float*)d_in[5];
    const float* Ub    = (const float*)d_in[6];
    float* out = (float*)d_out;

    // Workspace layout (f32 elements)
    float* ws = (float*)d_ws;
    const size_t xpre_elems = (size_t)T_DIM * B_DIM * GATES * H_DIM;
    float* xpre = ws;
    float* zx   = xpre + xpre_elems;
    float* zh   = zx + (size_t)GATES * B_DIM * I_DIM;
    float* hbuf = zh + (size_t)GATES * B_DIM * H_DIM;   // double-buffered h
    float* cbuf = hbuf + (size_t)2 * B_DIM * H_DIM;

    // 1) Dropout masks
    const int nzx = GATES * B_DIM * I_DIM;
    const int nzh = GATES * B_DIM * H_DIM;
    mask_kernel<<<(nzx + 255) / 256, 256, 0, stream>>>(ux, zx, nzx);
    mask_kernel<<<(nzh + 255) / 256, 256, 0, stream>>>(uh, zh, nzh);

    // 2) h0 = c0 = 0 (hbuf and cbuf are contiguous)
    const int nz = 3 * B_DIM * H_DIM;
    zero_kernel<<<(nz + 255) / 256, 256, 0, stream>>>(hbuf, nz);

    // 3) Time-parallel input projection (bulk WMMA GEMM with LDS-staged W)
    dim3 ggrid(H_DIM / 16, (T_DIM * B_DIM / 16) / (8 * ROWS_PER_WAVE), GATES);
    xpre_gemm_kernel<<<ggrid, 256, 0, stream>>>(input, zx, W, Wb, xpre);

    // 4) Serial recurrence: one fused launch per timestep
    const size_t BH = (size_t)B_DIM * H_DIM;
    for (int t = 0; t < T_DIM; ++t) {
        const float* h_in = hbuf + (size_t)(t & 1) * BH;
        float* h_out      = hbuf + (size_t)((t + 1) & 1) * BH;
        float* hn_t       = out + (size_t)t * BH;
        float* htd = nullptr;
        float* ctd = nullptr;
        if (t == T_DIM - 1) {
            htd = out + (size_t)T_DIM * BH;
            ctd = htd + BH;
        }
        lstm_step_kernel<<<dim3(H_DIM / 16), 128, 0, stream>>>(
            xpre + (size_t)t * B_DIM * GATES * H_DIM,
            zh, U, Ub, h_in, h_out, cbuf, hn_t, htd, ctd);
    }
}
